// Critic_48043504173686
// MI455X (gfx1250) — matrix operand order repaired and checked by reading, not compiled, not run on hardware
//
#include <hip/hip_runtime.h>
#include <hip/hip_bf16.h>

typedef __attribute__((ext_vector_type(16))) __bf16 v16bf;
typedef __attribute__((ext_vector_type(8)))  float  v8f;

#define LN_EPS 1e-5f
#define NEG_SLOPE 0.2f

// ---------------------------------------------------------------------------
// Fused GEMM (Y = leaky(LN(X @ W^T + bias)))  using bf16 WMMA with 3-term
// hi/lo split for ~fp32 accuracy.  X: [B, IN] row-major, W: [OUT, IN]
// row-major (so B-matrix column n == W row n, contiguous in K).
// Block: OUT threads = OUT/32 waves; each wave covers 32 output columns
// (two 16x16 tiles).  One block computes 16 rows.
// ---------------------------------------------------------------------------
template <int IN, int OUT>
__global__ __launch_bounds__(128) void mlp_ln_leaky(
    const float* __restrict__ X, const float* __restrict__ W,
    const float* __restrict__ bias, const float* __restrict__ gamma,
    const float* __restrict__ beta, float* __restrict__ Y) {
  const int m0   = blockIdx.x * 16;
  const int wave = threadIdx.x >> 5;
  const int lane = threadIdx.x & 31;
  const int lrow = lane & 15;       // row (A) or column (B/C) within tile
  const int kh   = lane >> 4;       // which K-half of the fragment this lane holds
  const int n0   = wave * 32;

  v8f acc[2] = {};

  for (int k0 = 0; k0 < IN; k0 += 32) {
    // ---- A fragment: 16x32 bf16, lanes 0-15 hold K {8kh+0..7, 16+8kh+0..7}
    const float* arow = X + (m0 + lrow) * IN + k0 + 8 * kh;
    v16bf ah, al;
#pragma unroll
    for (int v = 0; v < 8; ++v) {
      const int koff = (v < 4) ? (2 * v) : (16 + 2 * (v - 4));
      const float a0 = arow[koff];
      const float a1 = arow[koff + 1];
      const __bf16 h0 = (__bf16)a0, h1 = (__bf16)a1;
      ah[2 * v]     = h0;
      ah[2 * v + 1] = h1;
      al[2 * v]     = (__bf16)(a0 - (float)h0);
      al[2 * v + 1] = (__bf16)(a1 - (float)h1);
    }
    // ---- Two N-tiles per wave
#pragma unroll
    for (int t = 0; t < 2; ++t) {
      // B fragment: 32x16 bf16, lane holds column N=lrow, K {8kh+0..7, 16+8kh+0..7}
      const float* brow = W + (n0 + 16 * t + lrow) * IN + k0 + 8 * kh;
      v16bf bh, bl;
#pragma unroll
      for (int v = 0; v < 16; ++v) {
        const float b = brow[(v < 8) ? v : v + 8];
        const __bf16 hb = (__bf16)b;
        bh[v] = hb;
        bl[v] = (__bf16)(b - (float)hb);
      }
      acc[t] = __builtin_amdgcn_wmma_f32_16x16x32_bf16(false, ah, false, bh,
                                                       (short)0, acc[t], false, false);
      acc[t] = __builtin_amdgcn_wmma_f32_16x16x32_bf16(false, al, false, bh,
                                                       (short)0, acc[t], false, false);
      acc[t] = __builtin_amdgcn_wmma_f32_16x16x32_bf16(false, ah, false, bl,
                                                       (short)0, acc[t], false, false);
    }
  }

  // ---- Spill C tiles (+bias) to LDS: C layout lane l, vgpr r ->
  //      (M = r + 8*(l>>4), N = (l&15) + tile base)
  __shared__ float tile[16][OUT];
  __shared__ float mu_s[16], rs_s[16];
#pragma unroll
  for (int r = 0; r < 8; ++r) {
    const int m = r + 8 * kh;
    tile[m][n0 + lrow]      = acc[0][r] + bias[n0 + lrow];
    tile[m][n0 + 16 + lrow] = acc[1][r] + bias[n0 + 16 + lrow];
  }
  __syncthreads();

  // ---- Row statistics (16 rows; one thread each)
  if (threadIdx.x < 16) {
    float s = 0.f, s2 = 0.f;
    for (int n = 0; n < OUT; ++n) {
      const float v = tile[threadIdx.x][n];
      s += v;
      s2 += v * v;
    }
    const float mu  = s * (1.0f / OUT);
    const float var = s2 * (1.0f / OUT) - mu * mu;
    mu_s[threadIdx.x] = mu;
    rs_s[threadIdx.x] = rsqrtf(var + LN_EPS);
  }
  __syncthreads();

  // ---- Apply LN + leaky, write out
  for (int idx = threadIdx.x; idx < 16 * OUT; idx += blockDim.x) {
    const int m = idx / OUT, n = idx % OUT;
    float v = (tile[m][n] - mu_s[m]) * rs_s[m] * gamma[n] + beta[n];
    v = (v >= 0.f) ? v : NEG_SLOPE * v;
    Y[(m0 + m) * OUT + n] = v;
  }
}

// ---------------------------------------------------------------------------
// M = h3 @ T   (h3: [B,32], T: [32,80] K-major) -> M: [B,80]
// One wave per 16 rows, 5 N-tiles, single K=32 WMMA step (x3 for split).
// ---------------------------------------------------------------------------
__global__ __launch_bounds__(32) void proj_T(const float* __restrict__ H,
                                             const float* __restrict__ T,
                                             float* __restrict__ Mout) {
  const int m0   = blockIdx.x * 16;
  const int lane = threadIdx.x & 31;
  const int lrow = lane & 15;
  const int kh   = lane >> 4;

  const float* arow = H + (m0 + lrow) * 32 + 8 * kh;
  v16bf ah, al;
#pragma unroll
  for (int v = 0; v < 8; ++v) {
    const int koff = (v < 4) ? (2 * v) : (16 + 2 * (v - 4));
    const float a0 = arow[koff];
    const float a1 = arow[koff + 1];
    const __bf16 h0 = (__bf16)a0, h1 = (__bf16)a1;
    ah[2 * v]     = h0;
    ah[2 * v + 1] = h1;
    al[2 * v]     = (__bf16)(a0 - (float)h0);
    al[2 * v + 1] = (__bf16)(a1 - (float)h1);
  }

#pragma unroll
  for (int t = 0; t < 5; ++t) {
    v16bf bh, bl;
#pragma unroll
    for (int v = 0; v < 16; ++v) {
      const float b = T[(8 * kh + ((v < 8) ? v : v + 8)) * 80 + 16 * t + lrow];  // B[k][n] = T[k*80+n]
      const __bf16 hb = (__bf16)b;
      bh[v] = hb;
      bl[v] = (__bf16)(b - (float)hb);
    }
    v8f acc = {};
    acc = __builtin_amdgcn_wmma_f32_16x16x32_bf16(false, ah, false, bh,
                                                  (short)0, acc, false, false);
    acc = __builtin_amdgcn_wmma_f32_16x16x32_bf16(false, al, false, bh,
                                                  (short)0, acc, false, false);
    acc = __builtin_amdgcn_wmma_f32_16x16x32_bf16(false, ah, false, bl,
                                                  (short)0, acc, false, false);
#pragma unroll
    for (int r = 0; r < 8; ++r)
      Mout[(m0 + r + 8 * kh) * 80 + 16 * t + lrow] = acc[r];
  }
}

// ---------------------------------------------------------------------------
// o_b[i][o] = sum_j exp(-sum_k |M[i,o,k]-M[j,o,k]|) - 1
//
// Blocked 16x64: one block owns 16 i-rows; thread (ti,tj) keeps M[i] in 80
// VGPRs.  j processed in 64-row chunks double-buffered through LDS (rows
// padded to 84 floats -> conflict-free ds_load_b128 across the 16 tj lanes,
// broadcast across the 16 ti lanes).  Chunk c+1 is staged to registers while
// chunk c computes; chunk c+2 is prefetched (global_prefetch_b8).
// L2 traffic: 128 blocks x 640KB = ~82 MB (16x less than row-per-block).
// ---------------------------------------------------------------------------
#define PTI 16
#define PTJ 64
#define MROW 84  // padded LDS row stride (floats); 84*4B is 16B-aligned

__global__ __launch_bounds__(256) void pairwise_l1(const float* __restrict__ Mmat,
                                                   float* __restrict__ ob,
                                                   int B) {
  const int ti = threadIdx.x >> 4;  // 0..15  (i-row within block)
  const int tj = threadIdx.x & 15;  // 0..15  (j-lane within chunk)
  const int i  = blockIdx.x * PTI + ti;

  // own i-row in registers (16 tj-threads share it -> broadcast loads)
  float mi[80];
  {
    const float4* mrow = (const float4*)(Mmat + i * 80);
#pragma unroll
    for (int k = 0; k < 20; ++k) {
      const float4 v = mrow[k];
      mi[4 * k + 0] = v.x; mi[4 * k + 1] = v.y;
      mi[4 * k + 2] = v.z; mi[4 * k + 3] = v.w;
    }
  }

  __shared__ float buf[2][PTJ * MROW];
  const int NCH = B / PTJ;  // 32 chunks of 64 j-rows

  // Each chunk = 64*80 floats = 1280 float4; 256 threads -> 5 float4 each.
  float4 stg[5];
  auto stage_regs = [&](int c) {
#pragma unroll
    for (int q = 0; q < 5; ++q) {
      const int f = threadIdx.x + 256 * q;  // 0..1279
      stg[q] = ((const float4*)(Mmat + c * PTJ * 80))[f];
    }
  };
  auto regs_to_lds = [&](int slot) {
#pragma unroll
    for (int q = 0; q < 5; ++q) {
      const int f = threadIdx.x + 256 * q;
      const int j = f / 20, e = f % 20;
      ((float4*)&buf[slot][j * MROW])[e] = stg[q];
    }
  };

  float acc[5] = {0.f, 0.f, 0.f, 0.f, 0.f};

  stage_regs(0);
  regs_to_lds(0);
  __syncthreads();

  for (int c = 0; c < NCH; ++c) {
    if (c + 2 < NCH)
      __builtin_prefetch(Mmat + (c + 2) * PTJ * 80 + threadIdx.x * 20, 0, 0);
    if (c + 1 < NCH) stage_regs(c + 1);

    const float* base = buf[c & 1];
#pragma unroll
    for (int q = 0; q < 4; ++q) {
      const float4* mj = (const float4*)(base + (tj + 16 * q) * MROW);
#pragma unroll
      for (int o = 0; o < 5; ++o) {
        float l1 = 0.f;
#pragma unroll
        for (int p = 0; p < 4; ++p) {
          const float4 v = mj[o * 4 + p];
          l1 += fabsf(mi[o * 16 + 4 * p + 0] - v.x);
          l1 += fabsf(mi[o * 16 + 4 * p + 1] - v.y);
          l1 += fabsf(mi[o * 16 + 4 * p + 2] - v.z);
          l1 += fabsf(mi[o * 16 + 4 * p + 3] - v.w);
        }
        acc[o] += __expf(-l1);
      }
    }
    __syncthreads();
    if (c + 1 < NCH) {
      regs_to_lds((c + 1) & 1);
      __syncthreads();
    }
  }

  // reduce the 16 tj partials per i-row
  __shared__ float red[PTI][16][5];
#pragma unroll
  for (int o = 0; o < 5; ++o) red[ti][tj][o] = acc[o];
  __syncthreads();
  if (tj == 0) {
    float s[5] = {0.f, 0.f, 0.f, 0.f, 0.f};
    for (int t = 0; t < 16; ++t)
#pragma unroll
      for (int o = 0; o < 5; ++o) s[o] += red[ti][t][o];
#pragma unroll
    for (int o = 0; o < 5; ++o) ob[i * 5 + o] = s[o] - 1.0f;  // self term exp(0)
  }
}

// ---------------------------------------------------------------------------
// out[i] = [h3[i] ; o_b[i]] . Wf + bf
// ---------------------------------------------------------------------------
__global__ void final_head(const float* __restrict__ H3, const float* __restrict__ OB,
                           const float* __restrict__ Wf, const float* __restrict__ bf,
                           float* __restrict__ out, int B) {
  const int i = blockIdx.x * blockDim.x + threadIdx.x;
  if (i >= B) return;
  float s = bf[0];
#pragma unroll
  for (int k = 0; k < 32; ++k) s += H3[i * 32 + k] * Wf[k];
#pragma unroll
  for (int k = 0; k < 5; ++k) s += OB[i * 5 + k] * Wf[32 + k];
  out[i] = s;
}

// ---------------------------------------------------------------------------
extern "C" void kernel_launch(void* const* d_in, const int* in_sizes, int n_in,
                              void* d_out, int out_size, void* d_ws, size_t ws_size,
                              hipStream_t stream) {
  (void)in_sizes; (void)n_in; (void)out_size; (void)ws_size;
  const int B = 2048;

  const float* x   = (const float*)d_in[0];
  const float* W1  = (const float*)d_in[1];
  const float* b1  = (const float*)d_in[2];
  const float* g1  = (const float*)d_in[3];
  const float* be1 = (const float*)d_in[4];
  const float* W2  = (const float*)d_in[5];
  const float* b2  = (const float*)d_in[6];
  const float* g2  = (const float*)d_in[7];
  const float* be2 = (const float*)d_in[8];
  const float* W3  = (const float*)d_in[9];
  const float* b3  = (const float*)d_in[10];
  const float* g3  = (const float*)d_in[11];
  const float* be3 = (const float*)d_in[12];
  const float* T   = (const float*)d_in[13];
  const float* Wf  = (const float*)d_in[14];
  const float* bf  = (const float*)d_in[15];

  float* ws = (float*)d_ws;
  float* h1 = ws;                 // [2048,128]
  float* h2 = h1 + B * 128;       // [2048,64]
  float* h3 = h2 + B * 64;        // [2048,32]
  float* Mm = h3 + B * 32;        // [2048,80]
  float* ob = Mm + B * 80;        // [2048,5]

  mlp_ln_leaky<1024, 128><<<B / 16, 128, 0, stream>>>(x,  W1, b1, g1, be1, h1);
  mlp_ln_leaky<128,  64><<<B / 16,  64, 0, stream>>>(h1, W2, b2, g2, be2, h2);
  mlp_ln_leaky<64,   32><<<B / 16,  32, 0, stream>>>(h2, W3, b3, g3, be3, h3);
  proj_T<<<B / 16, 32, 0, stream>>>(h3, T, Mm);
  pairwise_l1<<<B / PTI, 256, 0, stream>>>(Mm, ob, B);
  final_head<<<(B + 255) / 256, 256, 0, stream>>>(h3, ob, Wf, bf, (float*)d_out, B);
}
